// AxialAttention_60782377173317
// MI455X (gfx1250) — matrix operand order repaired and checked by reading, not compiled, not run on hardware
//
#include <hip/hip_runtime.h>

typedef __attribute__((ext_vector_type(16))) _Float16 v16h;
typedef __attribute__((ext_vector_type(8)))  float    v8f;
typedef __attribute__((ext_vector_type(4)))  float    v4f;
typedef __attribute__((ext_vector_type(4)))  unsigned int v4u;

#define HEADS 8
#define EMB   128
#define SEQ   128
#define SW    136   // f16 row stride (halves): 128 + 8 pad (bank-conflict dodge)
#define SSW   132   // f32 row stride for S (floats)

union AFrag {
    v16h v;
    v4u  q[2];     // two 16-byte chunks
};

__device__ inline v4u zero4() {
    v4u z; z.x = 0u; z.y = 0u; z.z = 0u; z.w = 0u; return z;
}

// -----------------------------------------------------------------------------
// One workgroup (256 threads = 8 wave32) per (n, w) attention problem.
// Math trick: S = (y Wq^T)(y Wk^T)^T = (y G) y^T with G = Wq^T Wk (16x16),
// eliminating the K projection GEMM.
// All WMMAs issued in independent pairs/groups so the f16-WMMA 4-slot
// D-read hazard is filled with useful work instead of v_nops.
//
// LDS (dynamic, ~272 KB of 320 KB/WGP):
//   sG   : G^T as f16 rows [e'][d]         256 halves
//   sWv  : Wv as f16                       256 halves
//   sZ   : 32 zero halves (branchless zero fragments)
//   sWo  : Wo as f16                       128*SW halves
//   sY   : y[h][c] f16 (stays live all rounds)
//   sQG  : (y G)[h][c] f16
//   sVt  : V transposed [c][h] f16
//   sP   : unnormalized softmax probs [q][k] f16
//   sAO  : attention output [h][c] f16
//   sS   : scores f32 [q][SSW]
//   sbo  : bias f32 [128];  sInv : 1/rowsum f32 [128]
// -----------------------------------------------------------------------------
__global__ __launch_bounds__(256) void axial_attn_f16wmma(
    const float* __restrict__ x,
    const float* __restrict__ Wq, const float* __restrict__ Wk,
    const float* __restrict__ Wv, const float* __restrict__ Wo,
    const float* __restrict__ bo, float* __restrict__ out)
{
    extern __shared__ _Float16 smem[];
    _Float16* sG   = smem;                 // 256 halves
    _Float16* sWv  = sG  + 256;            // 256 halves
    _Float16* sZ   = sWv + 256;            // 32 halves of zeros
    _Float16* sWo  = sZ  + 32;             // 128*SW
    _Float16* sY   = sWo + 128 * SW;
    _Float16* sQG  = sY  + 128 * SW;
    _Float16* sVt  = sQG + 128 * SW;
    _Float16* sP   = sVt + 128 * SW;
    _Float16* sAO  = sP  + 128 * SW;
    float*    sS   = (float*)(sAO + 128 * SW);
    float*    sbo  = sS + 128 * SSW;
    float*    sInv = sbo + 128;

    const int tid  = threadIdx.x;
    const int wv   = tid >> 5;       // wave id 0..7
    const int lane = tid & 31;
    const int lo   = lane & 15;      // N / M index within fragment
    const int hi   = lane >> 4;      // half selector

    const int n = blockIdx.x >> 7;   // batch
    const int w = blockIdx.x & 127;  // width position (attention along H)
    const float* xb = x + ((size_t)n << 21) + (size_t)w;

    // ---------------- Phase 0: stage inputs into LDS ----------------
    sWv[tid & 255] = (_Float16)Wv[tid];            // tid < 256
    if (tid < 32)  sZ[tid] = (_Float16)0.0f;
    if (tid < EMB) sbo[tid] = bo[tid];
    // G^T[e'][d] = sum_e Wq[e][d] * Wk[e][e']  (f32, one element per thread)
    {
        const int d = tid & 15, ep = tid >> 4;
        float g = 0.f;
        #pragma unroll
        for (int e = 0; e < 16; ++e)
            g += Wq[e * 16 + d] * Wk[e * 16 + ep];
        sG[ep * 16 + d] = (_Float16)g;
    }
    for (int i = tid; i < EMB * EMB; i += 256)
        sWo[(i >> 7) * SW + (i & 127)] = (_Float16)Wo[i];
    // y[h][c] = x[n, c, h, w]; strided gather is L2-resident (x fits in 192MB L2)
    for (int i = tid; i < EMB * SEQ; i += 256) {
        const int c = i >> 7, h = i & 127;
        sY[h * SW + c] = (_Float16)xb[((size_t)c << 14) + (h << 7)];
    }
    __syncthreads();

    // ---------------- Phase 1: QG = y*G and Vt = (y*Wv^T)^T ----------------
    // One shared A fragment feeds TWO independent WMMAs (G and Wv) per head:
    // half the A loads, and the pair fills the D-read hazard slots.
    {
        const int ht = wv;                         // wave owns one 16-row strip
        AFrag bg, bv;
        {
            const _Float16* bp = sG + lo * 16 + hi * 8;
            bg.q[0] = *(const v4u*)bp;
            bg.q[1] = zero4();
            const _Float16* vp = sWv + lo * 16 + hi * 8;
            bv.q[0] = *(const v4u*)vp;
            bv.q[1] = zero4();
        }
        #pragma unroll
        for (int head = 0; head < HEADS; ++head) {
            AFrag a;
            a.q[0] = *(const v4u*)&sY[(ht * 16 + lo) * SW + head * 16 + hi * 8];
            a.q[1] = zero4();
            v8f acc0 = {}, acc1 = {};
            acc0 = __builtin_amdgcn_wmma_f32_16x16x32_f16(false, a.v, false, bg.v,
                                                          (short)0, acc0, false, false);
            acc1 = __builtin_amdgcn_wmma_f32_16x16x32_f16(false, a.v, false, bv.v,
                                                          (short)0, acc1, false, false);
            #pragma unroll
            for (int r = 0; r < 8; ++r)
                sQG[(ht * 16 + hi * 8 + r) * SW + head * 16 + lo] = (_Float16)acc0[r];
            #pragma unroll
            for (int r = 0; r < 8; ++r)   // store V transposed: Vt[e][seq]
                sVt[(head * 16 + lo) * SW + ht * 16 + hi * 8 + r] = (_Float16)acc1[r];
        }
    }
    __syncthreads();

    // ---------------- Phase 2: attention per head ----------------
    const float rscale = 0.08838834764831845f;   // 1/sqrt(EMBED=128)
    for (int head = 0; head < HEADS; ++head) {
        // S = QG * y^T : wave wv owns q-strip; A hoisted; k tiles in
        // independent pairs (two WMMAs in flight before any D read).
        {
            const int qt = wv;
            AFrag a;
            a.q[0] = *(const v4u*)&sQG[(qt * 16 + lo) * SW + head * 16 + hi * 8];
            a.q[1] = zero4();
            #pragma unroll
            for (int kt = 0; kt < 8; kt += 2) {
                AFrag b0, b1;
                const _Float16* bp0 = sY + ((kt + 0) * 16 + lo) * SW + head * 16 + hi * 8;
                const _Float16* bp1 = sY + ((kt + 1) * 16 + lo) * SW + head * 16 + hi * 8;
                b0.q[0] = *(const v4u*)bp0;  b0.q[1] = zero4();
                b1.q[0] = *(const v4u*)bp1;  b1.q[1] = zero4();
                v8f acc0 = {}, acc1 = {};
                acc0 = __builtin_amdgcn_wmma_f32_16x16x32_f16(false, a.v, false, b0.v,
                                                              (short)0, acc0, false, false);
                acc1 = __builtin_amdgcn_wmma_f32_16x16x32_f16(false, a.v, false, b1.v,
                                                              (short)0, acc1, false, false);
                #pragma unroll
                for (int r = 0; r < 8; ++r) {
                    sS[(qt * 16 + hi * 8 + r) * SSW + (kt + 0) * 16 + lo] = acc0[r];
                    sS[(qt * 16 + hi * 8 + r) * SSW + (kt + 1) * 16 + lo] = acc1[r];
                }
            }
        }
        __syncthreads();

        // softmax over k: one row/thread, v4f vectorized, 2 passes.
        // Probs stay UNNORMALIZED in f16; 1/sum applied in the PV epilogue.
        if (tid < SEQ) {
            const float* row = sS + tid * SSW;
            v4f m4 = *(const v4f*)row;
            for (int j = 4; j < SEQ; j += 4) {
                v4f t = *(const v4f*)(row + j);
                m4.x = fmaxf(m4.x, t.x); m4.y = fmaxf(m4.y, t.y);
                m4.z = fmaxf(m4.z, t.z); m4.w = fmaxf(m4.w, t.w);
            }
            const float mx = fmaxf(fmaxf(m4.x, m4.y), fmaxf(m4.z, m4.w));
            float sum = 0.f;
            _Float16* prow = sP + tid * SW;
            for (int j = 0; j < SEQ; j += 4) {
                v4f t = *(const v4f*)(row + j);
                const float p0 = __expf((t.x - mx) * rscale);
                const float p1 = __expf((t.y - mx) * rscale);
                const float p2 = __expf((t.z - mx) * rscale);
                const float p3 = __expf((t.w - mx) * rscale);
                sum += (p0 + p1) + (p2 + p3);
                prow[j]     = (_Float16)p0; prow[j + 1] = (_Float16)p1;
                prow[j + 2] = (_Float16)p2; prow[j + 3] = (_Float16)p3;
            }
            sInv[tid] = 1.0f / sum;
        }
        __syncthreads();

        // O = P * V : K=128 split into two interleaved accumulator chains
        // (k0 = 0,64 and 32,96) -> independent WMMAs hide chain latency;
        // merged with 8 VALU adds, scaled by 1/rowsum.
        {
            const int qt = wv;
            v8f accA = {}, accB = {};
            #pragma unroll
            for (int k0 = 0; k0 < SEQ; k0 += 64) {
                AFrag a0, b0, a1, b1;
                a0.q[0] = *(const v4u*)&sP[(qt * 16 + lo) * SW + k0 + hi * 8];
                a0.q[1] = *(const v4u*)&sP[(qt * 16 + lo) * SW + k0 + 16 + hi * 8];
                b0.q[0] = *(const v4u*)&sVt[(head * 16 + lo) * SW + k0 + hi * 8];
                b0.q[1] = *(const v4u*)&sVt[(head * 16 + lo) * SW + k0 + 16 + hi * 8];
                a1.q[0] = *(const v4u*)&sP[(qt * 16 + lo) * SW + k0 + 32 + hi * 8];
                a1.q[1] = *(const v4u*)&sP[(qt * 16 + lo) * SW + k0 + 48 + hi * 8];
                b1.q[0] = *(const v4u*)&sVt[(head * 16 + lo) * SW + k0 + 32 + hi * 8];
                b1.q[1] = *(const v4u*)&sVt[(head * 16 + lo) * SW + k0 + 48 + hi * 8];
                accA = __builtin_amdgcn_wmma_f32_16x16x32_f16(false, a0.v, false, b0.v,
                                                              (short)0, accA, false, false);
                accB = __builtin_amdgcn_wmma_f32_16x16x32_f16(false, a1.v, false, b1.v,
                                                              (short)0, accB, false, false);
            }
            #pragma unroll
            for (int r = 0; r < 8; ++r) {
                const int row = qt * 16 + hi * 8 + r;
                sAO[row * SW + head * 16 + lo] = (_Float16)((accA[r] + accB[r]) * sInv[row]);
            }
        }
        __syncthreads();
    }

    // ---------------- Phase 3: output projection + bias, write [n,c,h,w] ----------------
    // Wave wv owns output-channel tile ct = wv. B (Wo) hoisted per k0 across
    // 8 independent accumulators -> 8 back-to-back independent WMMAs.
    {
        const int ct = wv;
        v8f acc[8];
        #pragma unroll
        for (int i = 0; i < 8; ++i) { v8f z = {}; acc[i] = z; }
        #pragma unroll
        for (int k0 = 0; k0 < EMB; k0 += 32) {
            AFrag b;
            b.q[0] = *(const v4u*)&sWo[(ct * 16 + lo) * SW + k0 + hi * 8];
            b.q[1] = *(const v4u*)&sWo[(ct * 16 + lo) * SW + k0 + 16 + hi * 8];
            #pragma unroll
            for (int ht = 0; ht < 8; ++ht) {
                AFrag a;
                a.q[0] = *(const v4u*)&sAO[(ht * 16 + lo) * SW + k0 + hi * 8];
                a.q[1] = *(const v4u*)&sAO[(ht * 16 + lo) * SW + k0 + 16 + hi * 8];
                acc[ht] = __builtin_amdgcn_wmma_f32_16x16x32_f16(false, a.v, false, b.v,
                                                                 (short)0, acc[ht], false, false);
            }
        }
        const int cout = ct * 16 + lo;
        const float bias = sbo[cout];
        float* ob = out + (((size_t)n * EMB + cout) * SEQ) * 128 + w;
        #pragma unroll
        for (int ht = 0; ht < 8; ++ht) {
            #pragma unroll
            for (int r = 0; r < 8; ++r) {
                const int h = ht * 16 + hi * 8 + r;
                ob[(size_t)h * 128] = acc[ht][r] + bias;   // scatter; merges in L2
            }
        }
    }
}

extern "C" void kernel_launch(void* const* d_in, const int* in_sizes, int n_in,
                              void* d_out, int out_size, void* d_ws, size_t ws_size,
                              hipStream_t stream) {
    const float* x  = (const float*)d_in[0];
    const float* Wq = (const float*)d_in[1];
    const float* Wk = (const float*)d_in[2];
    const float* Wv = (const float*)d_in[3];
    const float* Wo = (const float*)d_in[4];
    const float* bo = (const float*)d_in[5];
    float* out = (float*)d_out;

    constexpr size_t SMEM =
        (size_t)(256 + 256 + 32 + 6 * 128 * SW) * sizeof(_Float16) +  // f16 buffers
        (size_t)128 * SSW * sizeof(float) +                           // scores
        (size_t)256 * sizeof(float);                                  // bias + inv
    (void)hipFuncSetAttribute((const void*)axial_attn_f16wmma,
                              hipFuncAttributeMaxDynamicSharedMemorySize,
                              (int)SMEM);
    dim3 grid(8 * 128);   // one workgroup per (n, w) problem
    dim3 block(256);      // 8 wave32s
    axial_attn_f16wmma<<<grid, block, SMEM, stream>>>(x, Wq, Wk, Wv, Wo, bo, out);
}
